// TriplanarDecoder_67585605370028
// MI455X (gfx1250) — matrix operand order repaired and checked by reading, not compiled, not run on hardware
//
#include <hip/hip_runtime.h>
#include <hip/hip_bf16.h>
#include <stdint.h>

typedef __attribute__((ext_vector_type(16))) _Float16 v16h;
typedef __attribute__((ext_vector_type(8)))  float    v8f;
typedef __attribute__((ext_vector_type(2)))  float    v2f;

#define C_CH     32
#define PLANE_W  512
#define PLANE_HW (512 * 512)
#define FEAT_STRIDE 72   // halves per feature row (64 cols + pad), even
#define ACT_STRIDE  40   // halves per act row (32 cols + pad), even

union FragU { v16h v; uint32_t u[8]; };

// A-matrix fragment (16xK tile, f16), rows = points, from row-major LDS tile.
// CDNA5 16-bit A layout: lanes 0-15 hold M=lane, K = {0..7, 16..23};
// lanes 16-31 hold M=lane-16, K = {8..15, 24..31}. Halves per VGPR are (k, k+1).
static __device__ __forceinline__ v16h load_a(const _Float16* base, int lane,
                                              int rowStride, int koff) {
    FragU f;
    const int m  = lane & 15;
    const int kb = (lane < 16) ? 0 : 8;
    const _Float16* row = base + m * rowStride + koff;
#pragma unroll
    for (int r = 0; r < 8; ++r) {
        const int k = ((r >> 2) * 16) + kb + ((2 * r) & 7);
        f.u[r] = *(const uint32_t*)(row + k);
    }
    return f.v;
}

// B-matrix fragment (32x16 tile of W, f16) from K-major (transposed) LDS weights.
// CDNA5 16-bit B layout: lanes 0-15 hold N=lane, K = {0..7, 16..23}; lanes 16-31 hold
// N=lane-16, K = {8..15, 24..31}. Halves per VGPR are (k, k+1) -> contiguous in wT.
static __device__ __forceinline__ v16h load_b(const _Float16* wT, int lane,
                                              int kStride, int ntile, int koff) {
    FragU f;
    const int n  = (lane & 15) + 16 * ntile;
    const int kb = (lane < 16) ? 0 : 8;
    const _Float16* col = wT + n * kStride + koff + kb;
#pragma unroll
    for (int r = 0; r < 8; ++r) {
        f.u[r] = *(const uint32_t*)(col + ((r >> 2) * 16) + 2 * (r & 3));
    }
    return f.v;
}

static __device__ __forceinline__ v8f wmma_f16(v16h a, v16h b, v8f c) {
    return __builtin_amdgcn_wmma_f32_16x16x32_f16(false, a, false, b,
                                                  (short)0, c, false, false);
}

// Apply bias + LeakyReLU to a D fragment (f32 16x16 C/D layout) and store the
// 16x16 half-tile into a row-major f16 act tile.
static __device__ __forceinline__ void act_store(v8f d, float bias, int lane,
                                                 int ntile, _Float16* act) {
    const int n  = (lane & 15) + 16 * ntile;
    const int m0 = (lane < 16) ? 0 : 8;
#pragma unroll
    for (int r = 0; r < 8; ++r) {
        float x = d[r] + bias;
        x = (x >= 0.0f) ? x : 0.01f * x;
        act[(m0 + r) * ACT_STRIDE + n] = (_Float16)x;
    }
}

// One bilinear border-clamped sample over all 32 channels, accumulated.
// The two x-corners are fetched with a single float2 load at xl=min(x0,510)
// (always in-bounds); the border clamp is folded into the 4 coefficients, so
// the channel loop is just 2 x b64 loads + 4 FMAs.
static __device__ __forceinline__ void tri_sample(const float* __restrict__ img,
                                                  float gx, float gy,
                                                  float* acc) {
    const float ix = (gx + 1.0f) * 0.5f * (float)(PLANE_W - 1);
    const float iy = (gy + 1.0f) * 0.5f * (float)(PLANE_W - 1);
    const float fx0 = floorf(ix), fy0 = floorf(iy);
    const float tx = ix - fx0, ty = iy - fy0;
    const float wnw = (1.0f - tx) * (1.0f - ty);
    const float wne = tx * (1.0f - ty);
    const float wsw = (1.0f - tx) * ty;
    const float wse = tx * ty;
    const int x0 = min(max((int)fx0, 0), PLANE_W - 1);
    const int x1 = min(max((int)fx0 + 1, 0), PLANE_W - 1);
    const int y0 = min(max((int)fy0, 0), PLANE_W - 1);
    const int y1 = min(max((int)fy0 + 1, 0), PLANE_W - 1);
    const int xl = min(x0, PLANE_W - 2);

    // Redistribute bilinear weights onto the two loaded texels (xl, xl+1).
    const float at = ((x0 == xl) ? wnw : 0.0f) + ((x1 == xl) ? wne : 0.0f);
    const float bt = (wnw + wne) - at;
    const float ab = ((x0 == xl) ? wsw : 0.0f) + ((x1 == xl) ? wse : 0.0f);
    const float bb = (wsw + wse) - ab;

    const float* r0 = img + y0 * PLANE_W + xl;
    const float* r1 = img + y1 * PLANE_W + xl;
#pragma unroll 8
    for (int c = 0; c < C_CH; ++c) {
        const v2f t = *(const v2f*)(r0 + c * PLANE_HW);   // global_load_b64
        const v2f u = *(const v2f*)(r1 + c * PLANE_HW);   // global_load_b64
        acc[c] += at * t.x + bt * t.y + ab * u.x + bb * u.y;
    }
}

__global__ __launch_bounds__(128)
void triplanar_decoder_kernel(const float* __restrict__ xy,
                              const float* __restrict__ xz,
                              const float* __restrict__ yz,
                              const float* __restrict__ coords,
                              const float* __restrict__ gauss,
                              const float* __restrict__ W0, const float* __restrict__ b0,
                              const float* __restrict__ W1, const float* __restrict__ b1,
                              const float* __restrict__ W2, const float* __restrict__ b2,
                              const float* __restrict__ W3, const float* __restrict__ b3,
                              float* __restrict__ out) {
    __shared__ _Float16 sFeat[128 * FEAT_STRIDE];       // 18,432 B
    __shared__ _Float16 sW[2048 + 1024 + 1024 + 512];   //  9,216 B (all K-major)
    __shared__ _Float16 sAct[4 * 16 * ACT_STRIDE];      //  5,120 B (per-wave scratch)

    const int tid   = threadIdx.x;
    const int gbase = blockIdx.x * 128;

    // ---- Stage 1: weights -> f16, transposed to K-major ([N][K]) in LDS ----
    _Float16* w0t = sW;              // [32][64]
    _Float16* w1t = sW + 2048;       // [32][32]
    _Float16* w2t = sW + 3072;       // [32][32]
    _Float16* w3t = sW + 4096;       // [16][32], cols >=4 zero-padded
    for (int i = tid; i < 2048; i += 128) {
        const int n = i >> 6, k = i & 63;
        w0t[n * 64 + k] = (_Float16)W0[k * 32 + n];
    }
    for (int i = tid; i < 1024; i += 128) {
        const int n = i >> 5, k = i & 31;
        w1t[n * 32 + k] = (_Float16)W1[k * 32 + n];
        w2t[n * 32 + k] = (_Float16)W2[k * 32 + n];
    }
    for (int i = tid; i < 512; i += 128) {
        const int n = i >> 5, k = i & 31;
        w3t[n * 32 + k] = (_Float16)((n < 4) ? W3[k * 4 + n] : 0.0f);
    }

    // ---- Stage 2: per-thread feature vector (triplanar + pos-enc) -> LDS ----
    {
        const int p = gbase + tid;
        const float cx = coords[p * 3 + 0];
        const float cy = coords[p * 3 + 1];
        const float cz = coords[p * 3 + 2];

        float acc[C_CH];
#pragma unroll
        for (int c = 0; c < C_CH; ++c) acc[c] = 0.0f;
        tri_sample(xy, cx, cy, acc);
        tri_sample(xz, cx, cz, acc);
        tri_sample(yz, cy, cz, acc);

        _Float16* frow = sFeat + tid * FEAT_STRIDE;
#pragma unroll
        for (int c = 0; c < C_CH; ++c) frow[c] = (_Float16)(acc[c] * (1.0f / 3.0f));

        const float px = cx * 6.28318530718f;
        const float py = cy * 6.28318530718f;
        const float pz = cz * 6.28318530718f;
#pragma unroll
        for (int c = 0; c < 16; ++c) {
            const float pr = px * gauss[c] + py * gauss[16 + c] + pz * gauss[32 + c];
            frow[32 + c] = (_Float16)__sinf(pr);
            frow[48 + c] = (_Float16)__cosf(pr);
        }
    }
    __syncthreads();

    // ---- Stage 3: WMMA MLP, 2 tiles of 16 points per wave ----
    const int wv   = tid >> 5;
    const int lane = tid & 31;
    const int nn   = lane & 15;
    _Float16* act  = sAct + wv * 16 * ACT_STRIDE;

    const float bb0l = b0[nn],      bb0h = b0[16 + nn];
    const float bb1l = b1[nn],      bb1h = b1[16 + nn];
    const float bb2l = b2[nn],      bb2h = b2[16 + nn];
    const float bb3  = b3[nn & 3];

#pragma unroll
    for (int t = 0; t < 2; ++t) {
        const int rb = wv * 32 + t * 16;     // local row base in sFeat

        // Layer 0: [16x64] @ [64x32]
        const v16h a0 = load_a(sFeat + rb * FEAT_STRIDE, lane, FEAT_STRIDE, 0);
        const v16h a1 = load_a(sFeat + rb * FEAT_STRIDE, lane, FEAT_STRIDE, 32);
        v8f d0 = {}, d1 = {};
        d0 = wmma_f16(a0, load_b(w0t, lane, 64, 0, 0),  d0);
        d0 = wmma_f16(a1, load_b(w0t, lane, 64, 0, 32), d0);
        d1 = wmma_f16(a0, load_b(w0t, lane, 64, 1, 0),  d1);
        d1 = wmma_f16(a1, load_b(w0t, lane, 64, 1, 32), d1);
        act_store(d0, bb0l, lane, 0, act);
        act_store(d1, bb0h, lane, 1, act);
        asm volatile("s_wait_dscnt 0x0" ::: "memory");

        // Layer 1: [16x32] @ [32x32]
        v16h x = load_a(act, lane, ACT_STRIDE, 0);
        d0 = v8f{}; d1 = v8f{};
        d0 = wmma_f16(x, load_b(w1t, lane, 32, 0, 0), d0);
        d1 = wmma_f16(x, load_b(w1t, lane, 32, 1, 0), d1);
        act_store(d0, bb1l, lane, 0, act);
        act_store(d1, bb1h, lane, 1, act);
        asm volatile("s_wait_dscnt 0x0" ::: "memory");

        // Layer 2: [16x32] @ [32x32]
        x = load_a(act, lane, ACT_STRIDE, 0);
        d0 = v8f{}; d1 = v8f{};
        d0 = wmma_f16(x, load_b(w2t, lane, 32, 0, 0), d0);
        d1 = wmma_f16(x, load_b(w2t, lane, 32, 1, 0), d1);
        act_store(d0, bb2l, lane, 0, act);
        act_store(d1, bb2h, lane, 1, act);
        asm volatile("s_wait_dscnt 0x0" ::: "memory");

        // Layer 3: [16x32] @ [32x16] (only n<4 valid), no activation
        x = load_a(act, lane, ACT_STRIDE, 0);
        v8f o = {};
        o = wmma_f16(x, load_b(w3t, lane, 32, 0, 0), o);

        if (nn < 4) {
            const int m0 = (lane < 16) ? 0 : 8;
#pragma unroll
            for (int r = 0; r < 8; ++r) {
                const int gp = gbase + rb + m0 + r;
                out[gp * 4 + nn] = o[r] + bb3;
            }
        }
    }
}

extern "C" void kernel_launch(void* const* d_in, const int* in_sizes, int n_in,
                              void* d_out, int out_size, void* d_ws, size_t ws_size,
                              hipStream_t stream) {
    (void)n_in; (void)out_size; (void)d_ws; (void)ws_size;
    const float* xy     = (const float*)d_in[0];
    const float* xz     = (const float*)d_in[1];
    const float* yz     = (const float*)d_in[2];
    const float* coords = (const float*)d_in[3];
    const float* gauss  = (const float*)d_in[4];
    const float* W0     = (const float*)d_in[5];
    const float* b0     = (const float*)d_in[6];
    const float* W1     = (const float*)d_in[7];
    const float* b1     = (const float*)d_in[8];
    const float* W2     = (const float*)d_in[9];
    const float* b2     = (const float*)d_in[10];
    const float* W3     = (const float*)d_in[11];
    const float* b3     = (const float*)d_in[12];
    float* out = (float*)d_out;

    const int nPoints = in_sizes[3] / 3;        // 2048*512 = 1,048,576
    const int blocks  = nPoints / 128;          // 8192
    triplanar_decoder_kernel<<<blocks, 128, 0, stream>>>(
        xy, xz, yz, coords, gauss, W0, b0, W1, b1, W2, b2, W3, b3, out);
}
